// Mamba_fusion_enhancement_module_79018808312433
// MI455X (gfx1250) — matrix operand order repaired and checked by reading, not compiled, not run on hardware
//
#include <hip/hip_runtime.h>
#include <hip/hip_bf16.h>

// ---------------------------------------------------------------- constants
#define D_MODEL 64
#define D_INNER 128
#define D_STATE 16
#define DT_RANK 4
#define NB      4
#define LTOK    16384            // 128*128 tokens per batch
#define NTOK    (NB * LTOK)
#define TCH     256              // scan chunk length
#define NCH     (LTOK / TCH)     // 64 chunks

typedef __attribute__((ext_vector_type(16))) _Float16 v16h;
typedef __attribute__((ext_vector_type(8)))  float    v8f;

// ---------------------------------------------------------------- CDNA5 async copy helpers
// GLOBAL_LOAD_ASYNC_TO_LDS_B128 (GV mode): VDST = per-lane LDS byte address,
// VADDR = 64-bit global address.  Tracked by ASYNCcnt (ISA 15.18.3 / 08_async_tensor).
__device__ __forceinline__ void async_g2l_b128(void* lds, const void* gptr) {
  unsigned loff = (unsigned)(unsigned long long)lds;   // flat LDS ptr low bits = LDS offset
  asm volatile("global_load_async_to_lds_b128 %0, %1, off"
               :: "v"(loff), "v"(gptr)
               : "memory");
}
__device__ __forceinline__ void async_wait0() {
  asm volatile("s_wait_asynccnt 0x0" ::: "memory");
}

// ---------------------------------------------------------------- fragment helpers
// A-matrix 16x32 f16 fragment, gathered from LDS row-major tile (stride in halves).
// Layout: lane = (m, g); VGPR j holds K = 16*g + 2*j (contiguous, matches packed B).
__device__ __forceinline__ v16h a_frag(const _Float16* row0, int strideH, int k0) {
  const int lane = threadIdx.x & 31;
  const int m = lane & 15, g = lane >> 4;
  v16h a;
#pragma unroll
  for (int j = 0; j < 8; ++j) {
    const int k = k0 + (g << 4) + (j << 1);
    a[2 * j]     = row0[m * strideH + k];
    a[2 * j + 1] = row0[m * strideH + k + 1];
  }
  return a;
}

// B-matrix fragment: prepacked in lane order, one contiguous 32B load per lane.
__device__ __forceinline__ v16h b_frag(const _Float16* pk, int tile) {
  const int lane = threadIdx.x & 31;
  return *reinterpret_cast<const v16h*>(pk + ((((size_t)tile << 5) + lane) << 4));
}

__device__ __forceinline__ float sigmoidf_(float v) { return 1.0f / (1.0f + __expf(-v)); }

// ---------------------------------------------------------------- weight packing
__global__ void __launch_bounds__(32)
k_pack_w(const float* __restrict__ W, _Float16* __restrict__ out, int N, int K) {
  const int tiles_k = K >> 5;
  const int tile = blockIdx.x;
  const int nt = tile / tiles_k, kt = tile % tiles_k;
  const int lane = threadIdx.x;
  const int n = nt * 16 + (lane & 15);
  const int g = lane >> 4;
  _Float16* dst = out + ((((size_t)tile << 5) + lane) << 4);
#pragma unroll
  for (int j = 0; j < 8; ++j) {
    const int k = (kt << 5) + (g << 4) + (j << 1);
    float v0 = (n < N) ? W[(size_t)n * K + k]     : 0.0f;
    float v1 = (n < N) ? W[(size_t)n * K + k + 1] : 0.0f;
    dst[2 * j]     = (_Float16)v0;
    dst[2 * j + 1] = (_Float16)v1;
  }
}

__global__ void __launch_bounds__(256)
k_aneg(const float* __restrict__ A_log, float* __restrict__ Aneg) {
  int i = blockIdx.x * 256 + threadIdx.x;
  if (i < D_INNER * D_STATE) Aneg[i] = -__expf(A_log[i]);
}

// ---------------------------------------------------------------- fusion + residual init
__global__ void __launch_bounds__(256)
k_fuse(const float* __restrict__ x, const float* __restrict__ y,
       float* __restrict__ fusion, float* __restrict__ tok) {
  size_t id = (size_t)blockIdx.x * 256 + threadIdx.x;   // over NB*64*LTOK, (b,c,t) order
  int bb = (int)(id / (64 * (size_t)LTOK));
  int rem = (int)(id % (64 * (size_t)LTOK));
  int c = rem / LTOK, t = rem % LTOK;
  float v = x[id] + y[id];
  size_t o = ((size_t)bb * LTOK + t) * 64 + c;          // token-major
  fusion[o] = v;
  tok[o]    = v;
}

// ---------------------------------------------------------------- front end of a block:
// t = in2out_w@f + b ; RMSNorm ; xz = in_proj_w@norm  ->  xc (d<128), zc (d>=128)
__global__ void __launch_bounds__(128)
k_front(const float* __restrict__ fusion,
        const _Float16* __restrict__ pk_in2out,   // 4x2 tiles
        const _Float16* __restrict__ pk_inproj,   // 16x2 tiles
        const float* __restrict__ in2out_b,
        const float* __restrict__ norm_w,
        float* __restrict__ xc, float* __restrict__ zc, int flip) {
  __shared__ _Float16 actA[64 * 72];
  __shared__ float    t2[64 * 68];
  __shared__ _Float16 actB[64 * 72];
  const int bb = blockIdx.x;
  const int tokBase = blockIdx.y * 64;
  const int tid = threadIdx.x;

  for (int idx = tid; idx < 64 * 64; idx += 128) {
    int c = idx & 63, tl = idx >> 6;
    int t = tokBase + tl;
    int ts = flip ? (LTOK - 1 - t) : t;
    actA[tl * 72 + c] = (_Float16)fusion[((size_t)bb * LTOK + ts) * 64 + c];
  }
  __syncthreads();

  const int w = tid >> 5, lane = tid & 31;
  const int n = lane & 15, g = lane >> 4;

  // GEMM1: 64->64 (K tiles 2), wave w owns tokens [w*16, w*16+16)
#pragma unroll
  for (int nt = 0; nt < 4; ++nt) {
    v8f c = {};
#pragma unroll
    for (int kt = 0; kt < 2; ++kt) {
      v16h a = a_frag(&actA[(w * 16) * 72], 72, kt * 32);
      v16h b = b_frag(pk_in2out, nt * 2 + kt);
      c = __builtin_amdgcn_wmma_f32_16x16x32_f16(false, a, false, b, (short)0, c, false, false);
    }
    float bias = in2out_b[nt * 16 + n];
#pragma unroll
    for (int r = 0; r < 8; ++r)
      t2[(w * 16 + r + 8 * g) * 68 + nt * 16 + n] = c[r] + bias;
  }
  __syncthreads();

  if (tid < 64) {
    float s = 0.f;
    for (int c = 0; c < 64; ++c) { float v = t2[tid * 68 + c]; s += v * v; }
    float scale = rsqrtf(s * (1.0f / 64.0f) + 1e-5f);
    for (int c = 0; c < 64; ++c)
      actB[tid * 72 + c] = (_Float16)(t2[tid * 68 + c] * scale * norm_w[c]);
  }
  __syncthreads();

  // GEMM2: 64->256
  for (int nt = 0; nt < 16; ++nt) {
    v8f c = {};
#pragma unroll
    for (int kt = 0; kt < 2; ++kt) {
      v16h a = a_frag(&actB[(w * 16) * 72], 72, kt * 32);
      v16h b = b_frag(pk_inproj, nt * 2 + kt);
      c = __builtin_amdgcn_wmma_f32_16x16x32_f16(false, a, false, b, (short)0, c, false, false);
    }
    const int e = nt * 16 + n;
#pragma unroll
    for (int r = 0; r < 8; ++r) {
      const int t = tokBase + w * 16 + r + 8 * g;
      if (e < D_INNER) xc[((size_t)bb * D_INNER + e) * LTOK + t] = c[r];
      else             zc[((size_t)bb * D_INNER + (e - D_INNER)) * LTOK + t] = c[r];
    }
  }
}

// ---------------------------------------------------------------- depthwise causal conv + silu
__global__ void __launch_bounds__(256)
k_conv(const float* __restrict__ xc, const float* __restrict__ cw,
       const float* __restrict__ cb, float* __restrict__ u, int dir) {
  const int t  = blockIdx.x * 256 + threadIdx.x;     // position in dir order
  const int d  = blockIdx.y;
  const int bb = blockIdx.z;
  const float w0 = cw[d * 4 + 0], w1 = cw[d * 4 + 1], w2 = cw[d * 4 + 2], w3 = cw[d * 4 + 3];
  const float* xr = xc + ((size_t)bb * D_INNER + d) * LTOK;
  float acc = cb[d];
#pragma unroll
  for (int k = 0; k < 4; ++k) {
    int p = t - 3 + k;
    float v = 0.0f;
    if (p >= 0) v = xr[dir ? (LTOK - 1 - p) : p];
    acc += (k == 0 ? w0 : k == 1 ? w1 : k == 2 ? w2 : w3) * v;
  }
  u[((size_t)bb * D_INNER + d) * LTOK + t] = acc * sigmoidf_(acc);
}

// ---------------------------------------------------------------- x_proj (WMMA) + dt_proj + softplus
__global__ void __launch_bounds__(128)
k_xproj(const float* __restrict__ u, const _Float16* __restrict__ pk_xp,  // 3x4 tiles (Npad=48)
        const float* __restrict__ dtw, const float* __restrict__ dtb,
        float* __restrict__ delta, float* __restrict__ Bm, float* __restrict__ Cm) {
  __shared__ _Float16 uH[64 * 136];
  __shared__ float    xd[64 * 52];
  const int bb = blockIdx.x;
  const int tokBase = blockIdx.y * 64;
  const int tid = threadIdx.x;

  for (int idx = tid; idx < 64 * 128; idx += 128) {
    int tl = idx & 63, d = idx >> 6;
    uH[tl * 136 + d] = (_Float16)u[((size_t)bb * D_INNER + d) * LTOK + tokBase + tl];
  }
  __syncthreads();

  const int w = tid >> 5, lane = tid & 31;
  const int n = lane & 15, g = lane >> 4;
#pragma unroll
  for (int nt = 0; nt < 3; ++nt) {
    v8f c = {};
#pragma unroll
    for (int kt = 0; kt < 4; ++kt) {
      v16h a = a_frag(&uH[(w * 16) * 136], 136, kt * 32);
      v16h b = b_frag(pk_xp, nt * 4 + kt);
      c = __builtin_amdgcn_wmma_f32_16x16x32_f16(false, a, false, b, (short)0, c, false, false);
    }
#pragma unroll
    for (int r = 0; r < 8; ++r)
      xd[(w * 16 + r + 8 * g) * 52 + nt * 16 + n] = c[r];
  }
  __syncthreads();

  const int tl = tid >> 1, half = tid & 1, tglob = tokBase + tl;
  const float r0 = xd[tl * 52 + 0], r1 = xd[tl * 52 + 1];
  const float r2 = xd[tl * 52 + 2], r3 = xd[tl * 52 + 3];
  for (int dd = 0; dd < 64; ++dd) {
    const int d = half * 64 + dd;
    float acc = dtb[d] + r0 * dtw[d * 4 + 0] + r1 * dtw[d * 4 + 1]
                       + r2 * dtw[d * 4 + 2] + r3 * dtw[d * 4 + 3];
    float sp = (acc > 20.f) ? acc : log1pf(__expf(acc));
    delta[((size_t)bb * D_INNER + d) * LTOK + tglob] = sp;
  }
  if (half == 0) {
    for (int nn = 0; nn < D_STATE; ++nn)
      Bm[((size_t)bb * D_STATE + nn) * LTOK + tglob] = xd[tl * 52 + DT_RANK + nn];
  } else {
    for (int nn = 0; nn < D_STATE; ++nn)
      Cm[((size_t)bb * D_STATE + nn) * LTOK + tglob] = xd[tl * 52 + DT_RANK + D_STATE + nn];
  }
}

// ---------------------------------------------------------------- chunked scan: pass 1 (summaries)
// delta/u subtiles stream into LDS with GLOBAL_LOAD_ASYNC_TO_LDS_B128, double buffered.
__global__ void __launch_bounds__(128)
k_scan1(const float* __restrict__ delta, const float* __restrict__ u,
        const float* __restrict__ Bm, const float* __restrict__ Aneg,
        float* __restrict__ Aprod, float* __restrict__ Hend) {
  __shared__ float Bs[16 * 260];          // [n][t], row-contiguous for b128 async
  __shared__ float bufD[2][128 * 36];     // [d][t] stride 36 (16B aligned rows)
  __shared__ float bufU[2][128 * 36];
  const int chunk = blockIdx.x, bb = blockIdx.y;
  const int d = threadIdx.x;
  const int base = chunk * TCH;

  // stage whole-chunk B rows: 16 rows x 256 floats = 1024 b128 transfers
#pragma unroll
  for (int q = 0; q < 8; ++q) {
    int li = q * 128 + d;
    int nn = li >> 6, tt = (li & 63) << 2;
    async_g2l_b128(&Bs[nn * 260 + tt],
                   &Bm[((size_t)bb * D_STATE + nn) * LTOK + base + tt]);
  }
  // stage first delta/u subtile (128 x 32)
#pragma unroll
  for (int q = 0; q < 8; ++q) {
    int li = q * 128 + d;
    int dd = li >> 3, tt = (li & 7) << 2;
    size_t go = ((size_t)bb * D_INNER + dd) * LTOK + base + tt;
    async_g2l_b128(&bufD[0][dd * 36 + tt], &delta[go]);
    async_g2l_b128(&bufU[0][dd * 36 + tt], &u[go]);
  }

  float A[16], h[16], ap[16];
#pragma unroll
  for (int nn = 0; nn < 16; ++nn) { A[nn] = Aneg[d * 16 + nn]; h[nn] = 0.f; ap[nn] = 1.f; }

  async_wait0();
  __syncthreads();

  for (int t0 = 0; t0 < TCH; t0 += 32) {
    const int cur = (t0 >> 5) & 1;
    if (t0 + 32 < TCH) {          // issue next subtile; overlaps with compute below
      const int P0 = base + t0 + 32;
#pragma unroll
      for (int q = 0; q < 8; ++q) {
        int li = q * 128 + d;
        int dd = li >> 3, tt = (li & 7) << 2;
        size_t go = ((size_t)bb * D_INNER + dd) * LTOK + P0 + tt;
        async_g2l_b128(&bufD[1 - cur][dd * 36 + tt], &delta[go]);
        async_g2l_b128(&bufU[1 - cur][dd * 36 + tt], &u[go]);
      }
    }
    const float* Dv = &bufD[cur][d * 36];
    const float* Uv = &bufU[cur][d * 36];
    for (int tt = 0; tt < 32; ++tt) {
      float dt = Dv[tt];
      float du = dt * Uv[tt];
      const float* Bt = &Bs[t0 + tt];
#pragma unroll
      for (int nn = 0; nn < 16; ++nn) {
        float e = __expf(dt * A[nn]);
        h[nn] = h[nn] * e + du * Bt[nn * 260];
        ap[nn] *= e;
      }
    }
    async_wait0();
    __syncthreads();
  }
#pragma unroll
  for (int nn = 0; nn < 16; ++nn) {
    size_t o = (((size_t)bb * NCH + chunk) * 128 + d) * 16 + nn;
    Aprod[o] = ap[nn];
    Hend[o]  = h[nn];
  }
}

// ---------------------------------------------------------------- pass 2: exclusive chunk prefix
__global__ void __launch_bounds__(256)
k_scan2(const float* __restrict__ Aprod, const float* __restrict__ Hend,
        float* __restrict__ Hin) {
  int id = blockIdx.x * 256 + threadIdx.x;     // NB*2048 lanes
  int bb = id >> 11, rest = id & 2047;
  float h = 0.f;
  for (int ch = 0; ch < NCH; ++ch) {
    size_t o = ((size_t)bb * NCH + ch) * 2048 + rest;
    Hin[o] = h;
    h = Aprod[o] * h + Hend[o];
  }
}

// ---------------------------------------------------------------- pass 3: replay + outputs + gate
__global__ void __launch_bounds__(128)
k_scan3(const float* __restrict__ delta, const float* __restrict__ u,
        const float* __restrict__ Bm, const float* __restrict__ Cm,
        const float* __restrict__ zc, const float* __restrict__ Aneg,
        const float* __restrict__ Dp, const float* __restrict__ Hin,
        float* __restrict__ ys, int dir) {
  __shared__ float Bs[16 * 260];
  __shared__ float Cs[16 * 260];
  __shared__ float bufD[2][128 * 36];
  __shared__ float bufU[2][128 * 36];
  __shared__ float bufZ[2][128 * 36];     // loaded in ascending global order; mirrored read for dir=1
  const int chunk = blockIdx.x, bb = blockIdx.y;
  const int d = threadIdx.x;
  const int base = chunk * TCH;
  const float Dd = Dp[d];

  // stage whole-chunk B and C rows
#pragma unroll
  for (int q = 0; q < 8; ++q) {
    int li = q * 128 + d;
    int nn = li >> 6, tt = (li & 63) << 2;
    size_t go = ((size_t)bb * D_STATE + nn) * LTOK + base + tt;
    async_g2l_b128(&Bs[nn * 260 + tt], &Bm[go]);
    async_g2l_b128(&Cs[nn * 260 + tt], &Cm[go]);
  }
  // stage first subtile
  {
    const int P0 = base;
    const int zb = dir ? (LTOK - 32 - P0) : P0;
#pragma unroll
    for (int q = 0; q < 8; ++q) {
      int li = q * 128 + d;
      int dd = li >> 3, tt = (li & 7) << 2;
      size_t row = ((size_t)bb * D_INNER + dd) * LTOK;
      async_g2l_b128(&bufD[0][dd * 36 + tt], &delta[row + P0 + tt]);
      async_g2l_b128(&bufU[0][dd * 36 + tt], &u[row + P0 + tt]);
      async_g2l_b128(&bufZ[0][dd * 36 + tt], &zc[row + zb + tt]);
    }
  }

  float A[16], h[16];
#pragma unroll
  for (int nn = 0; nn < 16; ++nn) {
    A[nn] = Aneg[d * 16 + nn];
    h[nn] = Hin[(((size_t)bb * NCH + chunk) * 128 + d) * 16 + nn];
  }

  async_wait0();
  __syncthreads();

  for (int t0 = 0; t0 < TCH; t0 += 32) {
    const int cur = (t0 >> 5) & 1;
    if (t0 + 32 < TCH) {
      const int P0 = base + t0 + 32;
      const int zb = dir ? (LTOK - 32 - P0) : P0;
#pragma unroll
      for (int q = 0; q < 8; ++q) {
        int li = q * 128 + d;
        int dd = li >> 3, tt = (li & 7) << 2;
        size_t row = ((size_t)bb * D_INNER + dd) * LTOK;
        async_g2l_b128(&bufD[1 - cur][dd * 36 + tt], &delta[row + P0 + tt]);
        async_g2l_b128(&bufU[1 - cur][dd * 36 + tt], &u[row + P0 + tt]);
        async_g2l_b128(&bufZ[1 - cur][dd * 36 + tt], &zc[row + zb + tt]);
      }
    }
    const float* Dv = &bufD[cur][d * 36];
    const float* Uv = &bufU[cur][d * 36];
    const float* Zv = &bufZ[cur][d * 36];
    for (int tt = 0; tt < 32; ++tt) {
      float dt = Dv[tt];
      float uv = Uv[tt];
      float du = dt * uv;
      const float* Bt = &Bs[t0 + tt];
      const float* Ct = &Cs[t0 + tt];
      float yv = 0.f;
#pragma unroll
      for (int nn = 0; nn < 16; ++nn) {
        float e = __expf(dt * A[nn]);
        h[nn] = h[nn] * e + du * Bt[nn * 260];
        yv += h[nn] * Ct[nn * 260];
      }
      yv += Dd * uv;
      float z = Zv[dir ? (31 - tt) : tt];
      ys[((size_t)bb * D_INNER + d) * LTOK + base + t0 + tt] = yv * (z * sigmoidf_(z));
    }
    async_wait0();
    __syncthreads();
  }
}

// ---------------------------------------------------------------- combine dirs + out_proj (WMMA) + residual acc
__global__ void __launch_bounds__(128)
k_combine(const float* __restrict__ yf, const float* __restrict__ yb,
          const _Float16* __restrict__ pk_out,   // 4x4 tiles
          float* __restrict__ tok, int flip) {
  __shared__ _Float16 yH[64 * 136];
  const int bb = blockIdx.x;
  const int tokBase = blockIdx.y * 64;
  const int tid = threadIdx.x;

  for (int idx = tid; idx < 64 * 128; idx += 128) {
    int tl = idx & 63, d = idx >> 6;
    int t = tokBase + tl;
    float v = 0.5f * (yf[((size_t)bb * D_INNER + d) * LTOK + t] +
                      yb[((size_t)bb * D_INNER + d) * LTOK + (LTOK - 1 - t)]);
    yH[tl * 136 + d] = (_Float16)v;
  }
  __syncthreads();

  const int w = tid >> 5, lane = tid & 31;
  const int n = lane & 15, g = lane >> 4;
#pragma unroll
  for (int nt = 0; nt < 4; ++nt) {
    v8f c = {};
#pragma unroll
    for (int kt = 0; kt < 4; ++kt) {
      v16h a = a_frag(&yH[(w * 16) * 136], 136, kt * 32);
      v16h b = b_frag(pk_out, nt * 4 + kt);
      c = __builtin_amdgcn_wmma_f32_16x16x32_f16(false, a, false, b, (short)0, c, false, false);
    }
#pragma unroll
    for (int r = 0; r < 8; ++r) {
      const int t = tokBase + w * 16 + r + 8 * g;
      const int td = flip ? (LTOK - 1 - t) : t;
      tok[((size_t)bb * LTOK + td) * 64 + nt * 16 + n] += c[r];
    }
  }
}

// ---------------------------------------------------------------- final LayerNorm + NCHW transpose
__global__ void __launch_bounds__(256)
k_norm(const float* __restrict__ tok, const float* __restrict__ lnw,
       const float* __restrict__ lnb, float* __restrict__ out) {
  int id = blockIdx.x * 256 + threadIdx.x;    // token id
  int bb = id / LTOK, t = id % LTOK;
  const float* row = tok + (size_t)id * 64;
  float s = 0.f, s2 = 0.f;
#pragma unroll 4
  for (int c = 0; c < 64; ++c) { float v = row[c]; s += v; s2 += v * v; }
  float mu  = s * (1.0f / 64.0f);
  float var = s2 * (1.0f / 64.0f) - mu * mu;
  float inv = rsqrtf(var + 64.0f);   // LN_EPS = 64.0 per reference
  for (int c = 0; c < 64; ++c)
    out[((size_t)bb * 64 + c) * LTOK + t] = (row[c] - mu) * inv * lnw[c] + lnb[c];
}

// ---------------------------------------------------------------- host orchestration
extern "C" void kernel_launch(void* const* d_in, const int* in_sizes, int n_in,
                              void* d_out, int out_size, void* d_ws, size_t ws_size,
                              hipStream_t stream) {
  (void)in_sizes; (void)n_in; (void)out_size; (void)ws_size;
  const float* x   = (const float*)d_in[0];
  const float* y   = (const float*)d_in[1];
  const float* lnw = (const float*)d_in[40];
  const float* lnb = (const float*)d_in[41];
  // nested param dicts flattened in sorted-key order, 19 arrays each
  enum { iA_log = 0, iA_log_b, iD, iD_b, iconv_b, iconv_b_b, iconv_w, iconv_w_b,
         idtproj_b, idtproj_b_b, idtproj_w, idtproj_w_b, iin2out_b, iin2out_w,
         iin_proj_w, inorm_w, iout_proj_w, ixproj_w, ixproj_w_b };
  auto PRM = [&](int blk, int k) -> const float* {
    return (const float*)d_in[(blk ? 21 : 2) + k];
  };

  // workspace carve
  char* base = (char*)d_ws;
  size_t off = 0;
  auto carve = [&](size_t bytes) -> char* {
    off = (off + 255) & ~(size_t)255;
    char* p = base + off;
    off += bytes;
    return p;
  };
  const size_t fTok  = (size_t)NTOK * 64 * sizeof(float);
  const size_t fChan = (size_t)NB * D_INNER * LTOK * sizeof(float);
  const size_t fSt   = (size_t)NB * D_STATE * LTOK * sizeof(float);
  const size_t fSum  = (size_t)NB * NCH * D_INNER * D_STATE * sizeof(float);

  float* fusion = (float*)carve(fTok);
  float* tok    = (float*)carve(fTok);
  float* xc     = (float*)carve(fChan);
  float* zc     = (float*)carve(fChan);
  float* u      = (float*)carve(fChan);
  float* delta  = (float*)carve(fChan);
  float* Bmb    = (float*)carve(fSt);
  float* Cmb    = (float*)carve(fSt);
  float* ys     = (float*)carve(2 * fChan);       // both directions
  float* Aprod  = (float*)carve(fSum);
  float* Hend   = (float*)carve(fSum);
  float* Hin    = (float*)carve(fSum);
  _Float16* pk_in2out = (_Float16*)carve(8  * 512 * sizeof(_Float16));
  _Float16* pk_inproj = (_Float16*)carve(32 * 512 * sizeof(_Float16));
  _Float16* pk_xp0    = (_Float16*)carve(12 * 512 * sizeof(_Float16));
  _Float16* pk_xp1    = (_Float16*)carve(12 * 512 * sizeof(_Float16));
  _Float16* pk_out    = (_Float16*)carve(16 * 512 * sizeof(_Float16));
  float* aneg0 = (float*)carve(D_INNER * D_STATE * sizeof(float));
  float* aneg1 = (float*)carve(D_INNER * D_STATE * sizeof(float));

  k_fuse<<<(unsigned)((size_t)NTOK * 64 / 256), 256, 0, stream>>>(x, y, fusion, tok);

  for (int blk = 0; blk < 2; ++blk) {
    k_pack_w<<<8,  32, 0, stream>>>(PRM(blk, iin2out_w),   pk_in2out, 64, 64);
    k_pack_w<<<32, 32, 0, stream>>>(PRM(blk, iin_proj_w),  pk_inproj, 256, 64);
    k_pack_w<<<12, 32, 0, stream>>>(PRM(blk, ixproj_w),    pk_xp0, 36, 128);
    k_pack_w<<<12, 32, 0, stream>>>(PRM(blk, ixproj_w_b),  pk_xp1, 36, 128);
    k_pack_w<<<16, 32, 0, stream>>>(PRM(blk, iout_proj_w), pk_out, 64, 128);
    k_aneg<<<8, 256, 0, stream>>>(PRM(blk, iA_log),   aneg0);
    k_aneg<<<8, 256, 0, stream>>>(PRM(blk, iA_log_b), aneg1);

    k_front<<<dim3(NB, LTOK / 64), 128, 0, stream>>>(
        fusion, pk_in2out, pk_inproj, PRM(blk, iin2out_b), PRM(blk, inorm_w),
        xc, zc, blk);

    for (int dir = 0; dir < 2; ++dir) {
      const float* cw  = PRM(blk, dir ? iconv_w_b   : iconv_w);
      const float* cb  = PRM(blk, dir ? iconv_b_b   : iconv_b);
      const float* dtw = PRM(blk, dir ? idtproj_w_b : idtproj_w);
      const float* dtb = PRM(blk, dir ? idtproj_b_b : idtproj_b);
      const float* an  = dir ? aneg1 : aneg0;
      const float* Dp  = PRM(blk, dir ? iD_b : iD);
      const _Float16* pkx = dir ? pk_xp1 : pk_xp0;
      float* ysd = ys + (size_t)dir * NB * D_INNER * LTOK;

      k_conv<<<dim3(LTOK / 256, D_INNER, NB), 256, 0, stream>>>(xc, cw, cb, u, dir);
      k_xproj<<<dim3(NB, LTOK / 64), 128, 0, stream>>>(u, pkx, dtw, dtb, delta, Bmb, Cmb);
      k_scan1<<<dim3(NCH, NB), 128, 0, stream>>>(delta, u, Bmb, an, Aprod, Hend);
      k_scan2<<<(NB * 2048) / 256, 256, 0, stream>>>(Aprod, Hend, Hin);
      k_scan3<<<dim3(NCH, NB), 128, 0, stream>>>(delta, u, Bmb, Cmb, zc, an, Dp, Hin, ysd, dir);
    }

    k_combine<<<dim3(NB, LTOK / 64), 128, 0, stream>>>(
        ys, ys + (size_t)NB * D_INNER * LTOK, pk_out, tok, blk);
  }

  k_norm<<<NTOK / 256, 256, 0, stream>>>(tok, lnw, lnb, (float*)d_out);
}
